// ABlock_36721970381534
// MI455X (gfx1250) — matrix operand order repaired and checked by reading, not compiled, not run on hardware
//
#include <hip/hip_runtime.h>
#include <hip/hip_bf16.h>

// ---------------------------------------------------------------------------
// ABlock for MI455X (gfx1250): LN -> attention (WMMA flash) -> MoE (WMMA GEMMs)
// GEMMs: v_wmma_f32_16x16x32_f16, LDS-tiled, async global->LDS staging (A),
// transposed LDS staging (B) so both A and B fragments are ds_load_b128 pairs.
// Attention K/V chunks staged with the Tensor Data Mover when available
// (tensor_load_to_lds + s_wait_tensorcnt), with OOB rows zero-filled by TDM.
// ---------------------------------------------------------------------------

typedef _Float16 half_t;
typedef __attribute__((ext_vector_type(16))) _Float16 v16h;
typedef __attribute__((ext_vector_type(8)))  float    v8f;
typedef __attribute__((ext_vector_type(4)))  unsigned int v4u;
typedef __attribute__((ext_vector_type(8)))  int      v8i;
typedef __attribute__((ext_vector_type(4)))  int      v4i;

#define NB   8          // batch
#define CCH  768        // channels
#define SSEQ 729        // tokens per batch (9^3)
#define MTOK (NB*SSEQ)  // 5832 tokens
#define MPAD 5888       // 46 * 128
#define HIDD 2048
#define NEXP 3
#define NHEAD 8
#define DHEAD 96
#define PERB (CCH*SSEQ) // 559872

// ---- CDNA5 feature guards (fallback keeps the build green) ----------------
#if defined(__AMDGCN__) && __has_builtin(__builtin_amdgcn_global_load_async_to_lds_b128) && __has_builtin(__builtin_amdgcn_s_wait_asynccnt)
#define HAVE_ASYNC_LDS 1
typedef __attribute__((address_space(1))) v4i as1v4i;
typedef __attribute__((address_space(3))) v4i as3v4i;
#else
#define HAVE_ASYNC_LDS 0
#endif

#if defined(__AMDGCN__) && __has_builtin(__builtin_amdgcn_tensor_load_to_lds) && __has_builtin(__builtin_amdgcn_s_wait_tensorcnt)
#define HAVE_TDM 1
#else
#define HAVE_TDM 0
#endif

#if HAVE_TDM
__device__ inline unsigned lds_off_u32(const void* p) {
  return (unsigned)(unsigned long long)(__attribute__((address_space(3))) const void*)p;
}

// 2-D tile DMA: global (row-major, strided) -> LDS (packed). data_size = 2B.
// Rows/cols beyond (tdim0, tdim1) are zero-filled by the TDM.
__device__ inline void tdm_load_2d(unsigned lds_off, const void* gptr,
                                   unsigned tdim0, unsigned tdim1,
                                   unsigned tile0, unsigned tile1,
                                   unsigned long long stride0_elems) {
  unsigned long long ga = (unsigned long long)(size_t)gptr;
  v4u g0;
  g0[0] = 1u;                                          // count=1 (user D#)
  g0[1] = lds_off;                                     // lds_addr (bytes)
  g0[2] = (unsigned)(ga & 0xffffffffu);                // global_addr[31:0]
  g0[3] = (unsigned)((ga >> 32) & 0x01ffffffu) | (2u << 30);  // [56:32] | type=2
  v8i g1;
  g1[0] = (int)(1u << 16);                             // wg_mask=0, data_size=1 (2B)
  g1[1] = (int)((tdim0 & 0xffffu) << 16);              // tensor_dim0[15:0]
  g1[2] = (int)((tdim0 >> 16) | ((tdim1 & 0xffffu) << 16));
  g1[3] = (int)((tdim1 >> 16) | ((tile0 & 0xffffu) << 16));
  g1[4] = (int)(tile1 & 0xffffu);                      // tile_dim1; tile_dim2=0
  g1[5] = (int)(unsigned)(stride0_elems & 0xffffffffu);
  g1[6] = (int)(unsigned)((stride0_elems >> 32) & 0xffffu);  // dim1_stride=0
  g1[7] = 0;
  v4i z4 = {0, 0, 0, 0};
#if __clang_major__ >= 23
  v8i z8 = {0, 0, 0, 0, 0, 0, 0, 0};
  __builtin_amdgcn_tensor_load_to_lds(g0, g1, z4, z4, z8, 0);
#else
  __builtin_amdgcn_tensor_load_to_lds(g0, g1, z4, z4, 0);
#endif
}
#endif  // HAVE_TDM

// ------------------------------ LayerNorm ----------------------------------

__global__ __launch_bounds__(256) void ln_stats_kernel(const float* __restrict__ x,
                                                       float* __restrict__ stats) {
  int b = blockIdx.x;
  const float* p = x + (size_t)b * PERB;
  float s = 0.f, ss = 0.f;
  for (int i = threadIdx.x; i < PERB; i += 256) {
    float v = p[i];
    s += v; ss += v * v;
  }
  __shared__ float sh[256];
  __shared__ float sh2[256];
  sh[threadIdx.x] = s; sh2[threadIdx.x] = ss;
  __syncthreads();
  for (int o = 128; o > 0; o >>= 1) {
    if (threadIdx.x < o) { sh[threadIdx.x] += sh[threadIdx.x + o]; sh2[threadIdx.x] += sh2[threadIdx.x + o]; }
    __syncthreads();
  }
  if (threadIdx.x == 0) {
    float inv_n = 1.0f / (float)PERB;
    float mean = sh[0] * inv_n;
    float var  = sh2[0] * inv_n - mean * mean;
    stats[b * 2 + 0] = mean;
    stats[b * 2 + 1] = rsqrtf(var + 1e-5f);
  }
}

__global__ __launch_bounds__(256) void ln_apply_kernel(const float* __restrict__ x,
                                                       const float* __restrict__ lnw,
                                                       const float* __restrict__ lnb,
                                                       const float* __restrict__ stats,
                                                       half_t* __restrict__ tokH) {
  size_t i = (size_t)blockIdx.x * 256 + threadIdx.x;
  if (i >= (size_t)NB * PERB) return;
  int b = (int)(i / PERB);
  int rem = (int)(i % PERB);
  int c = rem / SSEQ;
  int sp = rem % SSEQ;
  float mean = stats[b * 2 + 0];
  float inv  = stats[b * 2 + 1];
  float y = (x[i] - mean) * inv * lnw[rem] + lnb[rem];
  tokH[((size_t)(b * SSEQ + sp)) * CCH + c] = (half_t)y;
}

// ------------------------------ f32 -> f16 cast ----------------------------

__global__ __launch_bounds__(256) void cast_f32_f16_kernel(const float* __restrict__ in,
                                                           half_t* __restrict__ out,
                                                           size_t n) {
  size_t i = (size_t)blockIdx.x * 256 + threadIdx.x;
  size_t stride = (size_t)gridDim.x * 256;
  for (; i < n; i += stride) out[i] = (half_t)in[i];
}

// ------------------------------ Generic WMMA GEMM --------------------------
// D(MxN) = A(MxK, f16, row-major) @ B(KxN, f16, row-major); f32 accumulate.
// Block tile 128x64; 8 waves in 4x2; wave tile 32x32; K step 32.
// A: async global->LDS (row-major). B: staged transposed so every fragment
// read on both sides is two contiguous ds_load_b128.

template<bool OUT_HALF>
__global__ __launch_bounds__(256) void gemm_f16_kernel(const half_t* __restrict__ A,
                                                       const half_t* __restrict__ B,
                                                       void* __restrict__ Dv,
                                                       int M, int N, int K) {
  __shared__ half_t As[128 * 40];  // 128 rows x 32 (pad to 40)
  __shared__ half_t Bt[64 * 40];   // 64 cols x 32 k (pad to 40), transposed
  const int tid  = threadIdx.x;
  const int lane = tid & 31;
  const int wave = tid >> 5;
  const int wm = wave >> 1, wn = wave & 1;
  const int l2 = lane & 15, hi = lane >> 4;
  const int rowBase = blockIdx.y * 128;
  const int colBase = blockIdx.x * 64;

  v8f zero = {};
  v8f acc[2][2];
  acc[0][0] = zero; acc[0][1] = zero; acc[1][0] = zero; acc[1][1] = zero;

  for (int k0 = 0; k0 < K; k0 += 32) {
    // ---- stage A tile (128x32) ----
#pragma unroll
    for (int it = 0; it < 2; ++it) {
      int vec = tid + it * 256;
      int r = vec >> 2;
      int c8 = (vec & 3) * 8;
      const half_t* src = A + (size_t)(rowBase + r) * K + k0 + c8;
      half_t* dst = &As[r * 40 + c8];
#if HAVE_ASYNC_LDS
      __builtin_amdgcn_global_load_async_to_lds_b128((as1v4i*)src, (as3v4i*)dst, 0, 0);
#else
      *reinterpret_cast<uint4*>(dst) = *reinterpret_cast<const uint4*>(src);
#endif
    }
    // ---- stage B tile (32x64) transposed into Bt[col][k] ----
    {
      int r = tid >> 3;             // k row 0..31
      int c8 = (tid & 7) * 8;       // column group
      uint4 d = *reinterpret_cast<const uint4*>(B + (size_t)(k0 + r) * N + colBase + c8);
      const half_t* hp = reinterpret_cast<const half_t*>(&d);
#pragma unroll
      for (int j = 0; j < 8; ++j) Bt[(c8 + j) * 40 + r] = hp[j];
    }
    if (k0 + 32 < K) {
      __builtin_prefetch(A + (size_t)(rowBase + (tid >> 1)) * K + k0 + 32, 0, 1);
      __builtin_prefetch(B + (size_t)(k0 + 32 + (tid >> 3)) * N + colBase, 0, 1);
    }
#if HAVE_ASYNC_LDS
    __builtin_amdgcn_s_wait_asynccnt(0);
#endif
    __syncthreads();

    // A fragments (16x32): lane l2 -> row; contiguous K {hi*16..hi*16+15}
    v16h af[2];
#pragma unroll
    for (int i = 0; i < 2; ++i) {
      const half_t* p = &As[(wm * 32 + i * 16 + l2) * 40];
#pragma unroll
      for (int j = 0; j < 4; ++j) { int k = hi * 16 + 2 * j;     af[i][2*j]   = p[k]; af[i][2*j+1]   = p[k+1]; }
#pragma unroll
      for (int j = 0; j < 4; ++j) { int k = hi * 16 + 8 + 2 * j; af[i][8+2*j] = p[k]; af[i][8+2*j+1] = p[k+1]; }
    }
    // B fragments (32x16): lane l2 -> column; contiguous 16 halfs per lane
    v16h bf[2];
#pragma unroll
    for (int j = 0; j < 2; ++j) {
      const half_t* p = &Bt[(wn * 32 + j * 16 + l2) * 40 + hi * 16];
#pragma unroll
      for (int t = 0; t < 16; ++t) bf[j][t] = p[t];
    }
#pragma unroll
    for (int i = 0; i < 2; ++i)
#pragma unroll
      for (int j = 0; j < 2; ++j)
        acc[i][j] = __builtin_amdgcn_wmma_f32_16x16x32_f16(
            false, af[i], false, bf[j], (short)0, acc[i][j], false, false);
    __syncthreads();
  }

#pragma unroll
  for (int i = 0; i < 2; ++i)
#pragma unroll
    for (int j = 0; j < 2; ++j) {
      int col = colBase + wn * 32 + j * 16 + l2;
#pragma unroll
      for (int r = 0; r < 8; ++r) {
        int row = rowBase + wm * 32 + i * 16 + r + hi * 8;
        if (row < M) {
          float v = acc[i][j][r];
          if (OUT_HALF) ((half_t*)Dv)[(size_t)row * N + col] = (half_t)v;
          else          ((float*)Dv)[(size_t)row * N + col]  = v;
        }
      }
    }
}

// ------------------------------ Flash attention ----------------------------
// grid (46 m-tiles, 8 heads, 8 batches); 1 wave / block; 16 query rows / wave.
// K/V chunks (32x96 f16) staged by the Tensor Data Mover when available.

__global__ __launch_bounds__(32) void attn_kernel(const half_t* __restrict__ qkv,
                                                  half_t* __restrict__ oH) {
  const int mt = blockIdx.x, h = blockIdx.y, b = blockIdx.z;
  const int lane = threadIdx.x;
  const int l2 = lane & 15, hi = lane >> 4;

  __shared__ half_t kt[32 * 96];
  __shared__ half_t vt[32 * 96];
  __shared__ half_t pt[16 * 32];

  const half_t* qbase = qkv + (size_t)(b * SSEQ) * 2304 + h * DHEAD;
  const half_t* kbase = qbase + 768;
  const half_t* vbase = qbase + 1536;

  // Q A-fragments (3 K-chunks of 32 covering dh=96), kept in registers
  v16h qa[3];
  {
    const half_t* qr = qbase + (size_t)(mt * 16 + l2) * 2304;
#pragma unroll
    for (int d0 = 0; d0 < 3; ++d0) {
#pragma unroll
      for (int j = 0; j < 4; ++j) { int k = d0*32 + hi*16 + 2*j;     qa[d0][2*j]   = qr[k]; qa[d0][2*j+1]   = qr[k+1]; }
#pragma unroll
      for (int j = 0; j < 4; ++j) { int k = d0*32 + hi*16 + 8 + 2*j; qa[d0][8+2*j] = qr[k]; qa[d0][8+2*j+1] = qr[k+1]; }
    }
  }

  v8f zero = {};
  v8f o[6];
#pragma unroll
  for (int f = 0; f < 6; ++f) o[f] = zero;
  float mrun[8], lrun[8];
#pragma unroll
  for (int r = 0; r < 8; ++r) { mrun[r] = -3.0e30f; lrun[r] = 0.f; }

  for (int ch = 0; ch < 23; ++ch) {
    const int kv0 = ch * 32;
#if HAVE_TDM
    // WAR fence: make sure prior-chunk LDS reads retired before DMA overwrites
    asm volatile("s_wait_dscnt 0x0" ::: "memory");
    {
      unsigned rows_left = (unsigned)(SSEQ - kv0);  // TDM zero-fills rows beyond
      tdm_load_2d(lds_off_u32(kt), kbase + (size_t)kv0 * 2304,
                  DHEAD, rows_left, DHEAD, 32, 2304ull);
      tdm_load_2d(lds_off_u32(vt), vbase + (size_t)kv0 * 2304,
                  DHEAD, rows_left, DHEAD, 32, 2304ull);
    }
    __builtin_amdgcn_s_wait_tensorcnt(0);
    __syncthreads();
#else
    // stage K/V chunk (32x96 halfs each), zero beyond s=729
    for (int vec = lane; vec < 32 * 12; vec += 32) {
      int r = vec / 12;
      int c8 = (vec % 12) * 8;
      int krow = kv0 + r;
      uint4 kz = {0, 0, 0, 0}, vz = {0, 0, 0, 0};
      if (krow < SSEQ) {
        kz = *reinterpret_cast<const uint4*>(kbase + (size_t)krow * 2304 + c8);
        vz = *reinterpret_cast<const uint4*>(vbase + (size_t)krow * 2304 + c8);
      }
      *reinterpret_cast<uint4*>(&kt[r * 96 + c8]) = kz;
      *reinterpret_cast<uint4*>(&vt[r * 96 + c8]) = vz;
    }
    __syncthreads();
#endif

    // S = Q @ K^T for this 16x32 chunk (two 16x16 C tiles)
    v8f s[2];
    s[0] = zero; s[1] = zero;
#pragma unroll
    for (int g = 0; g < 2; ++g) {
#pragma unroll
      for (int d0 = 0; d0 < 3; ++d0) {
        v16h bfr;
        const half_t* kc = &kt[(g * 16 + l2) * 96 + d0 * 32];
#pragma unroll
        for (int t = 0; t < 8; ++t) { int k = hi * 16 + 2 * t; bfr[2*t] = kc[k]; bfr[2*t+1] = kc[k+1]; }
        s[g] = __builtin_amdgcn_wmma_f32_16x16x32_f16(
            false, qa[d0], false, bfr, (short)0, s[g], false, false);
      }
    }

    // online softmax (row m lives in VGPR r across one 16-lane half)
    float scale[8];
#pragma unroll
    for (int r = 0; r < 8; ++r) {
      int n0 = kv0 + l2;
      int n1 = kv0 + 16 + l2;
      float a0 = (n0 < SSEQ) ? s[0][r] : -3.0e30f;
      float a1 = (n1 < SSEQ) ? s[1][r] : -3.0e30f;
      float m = fmaxf(a0, a1);
      for (int off = 1; off < 16; off <<= 1) m = fmaxf(m, __shfl_xor(m, off, 32));
      float mn = fmaxf(mrun[r], m);
      float sc = __expf(mrun[r] - mn);
      float p0 = __expf(a0 - mn);
      float p1 = __expf(a1 - mn);
      float ps = p0 + p1;
      for (int off = 1; off < 16; off <<= 1) ps += __shfl_xor(ps, off, 32);
      lrun[r] = lrun[r] * sc + ps;
      mrun[r] = mn;
      scale[r] = sc;
      s[0][r] = p0; s[1][r] = p1;
    }
#pragma unroll
    for (int f = 0; f < 6; ++f)
#pragma unroll
      for (int r = 0; r < 8; ++r) o[f][r] *= scale[r];

    // P (C layout) -> LDS -> A fragment for P @ V
    __syncthreads();
#pragma unroll
    for (int r = 0; r < 8; ++r) {
      int row = r + hi * 8;
      pt[row * 32 + l2]      = (half_t)s[0][r];
      pt[row * 32 + 16 + l2] = (half_t)s[1][r];
    }
    __syncthreads();
    v16h pa;
    {
      const half_t* pr = &pt[l2 * 32];
#pragma unroll
      for (int j = 0; j < 4; ++j) { int k = hi*16 + 2*j;     pa[2*j]   = pr[k]; pa[2*j+1]   = pr[k+1]; }
#pragma unroll
      for (int j = 0; j < 4; ++j) { int k = hi*16 + 8 + 2*j; pa[8+2*j] = pr[k]; pa[8+2*j+1] = pr[k+1]; }
    }
#pragma unroll
    for (int f = 0; f < 6; ++f) {
      v16h bfr;
      const half_t* vc = &vt[f * 16 + l2];
#pragma unroll
      for (int t = 0; t < 8; ++t) { int k = hi * 16 + 2 * t; bfr[2*t] = vc[k * 96]; bfr[2*t+1] = vc[(k + 1) * 96]; }
      o[f] = __builtin_amdgcn_wmma_f32_16x16x32_f16(
          false, pa, false, bfr, (short)0, o[f], false, false);
    }
    __syncthreads();
  }

#pragma unroll
  for (int f = 0; f < 6; ++f)
#pragma unroll
    for (int r = 0; r < 8; ++r) {
      int srow = mt * 16 + r + hi * 8;
      if (srow < SSEQ) {
        float v = o[f][r] / lrun[r];
        oH[((size_t)(b * SSEQ + srow)) * CCH + h * DHEAD + f * 16 + l2] = (half_t)v;
      }
    }
}

// ------------------------------ residual + token cast ----------------------

__global__ __launch_bounds__(256) void residual_attn_kernel(const float* __restrict__ x,
                                                            const float* __restrict__ projO,
                                                            float* __restrict__ x2,
                                                            half_t* __restrict__ tok2H) {
  size_t i = (size_t)blockIdx.x * 256 + threadIdx.x;
  if (i >= (size_t)NB * PERB) return;
  int b = (int)(i / PERB);
  int rem = (int)(i % PERB);
  int c = rem / SSEQ;
  int sp = rem % SSEQ;
  float v = x[i] + projO[((size_t)(b * SSEQ + sp)) * CCH + c];
  x2[i] = v;
  tok2H[((size_t)(b * SSEQ + sp)) * CCH + c] = (half_t)v;
}

// ------------------------------ router: softmax + top-2 --------------------

__global__ __launch_bounds__(128) void router_kernel(const float* __restrict__ x2,
                                                     const float* __restrict__ rw,
                                                     float* __restrict__ wfull) {
  int tok = blockIdx.x * 4 + (threadIdx.x >> 5);
  if (tok >= MTOK) return;
  int lane = threadIdx.x & 31;
  int b = tok / SSEQ, sp = tok % SSEQ;
  const float* xb = x2 + (size_t)b * PERB + sp;  // stride SSEQ per channel
  float a0 = 0.f, a1 = 0.f, a2 = 0.f;
  for (int c = lane; c < CCH; c += 32) {
    float t = xb[(size_t)c * SSEQ];
    a0 += t * rw[c * 3 + 0];
    a1 += t * rw[c * 3 + 1];
    a2 += t * rw[c * 3 + 2];
  }
  for (int off = 1; off < 32; off <<= 1) {
    a0 += __shfl_xor(a0, off, 32);
    a1 += __shfl_xor(a1, off, 32);
    a2 += __shfl_xor(a2, off, 32);
  }
  if (lane == 0) {
    float m = fmaxf(a0, fmaxf(a1, a2));
    float p[3] = {__expf(a0 - m), __expf(a1 - m), __expf(a2 - m)};
    float s = p[0] + p[1] + p[2];
    p[0] /= s; p[1] /= s; p[2] /= s;
    int i0 = 0;
    if (p[1] > p[i0]) i0 = 1;
    if (p[2] > p[i0]) i0 = 2;
    int i1 = (i0 == 0) ? 1 : 0;
    for (int i = 0; i < 3; ++i) if (i != i0 && p[i] > p[i1]) i1 = i;
    float w = p[i0] + p[i1];
    float outw[3] = {0.f, 0.f, 0.f};
    outw[i0] = p[i0] / w;
    outw[i1] = p[i1] / w;
    wfull[tok * 3 + 0] = outw[0];
    wfull[tok * 3 + 1] = outw[1];
    wfull[tok * 3 + 2] = outw[2];
  }
}

// ------------------------------ SwiGLU ------------------------------------

__global__ __launch_bounds__(256) void silu_mul_kernel(half_t* __restrict__ g,
                                                       const half_t* __restrict__ u,
                                                       size_t n) {
  size_t i = (size_t)blockIdx.x * 256 + threadIdx.x;
  size_t stride = (size_t)gridDim.x * 256;
  for (; i < n; i += stride) {
    float gv = (float)g[i];
    float uv = (float)u[i];
    g[i] = (half_t)(gv / (1.f + __expf(-gv)) * uv);
  }
}

// ------------------------------ MoE combine + residual ---------------------

__global__ __launch_bounds__(256) void moe_combine_kernel(const float* __restrict__ x2,
                                                          const float* __restrict__ eo,
                                                          const float* __restrict__ wfull,
                                                          float* __restrict__ out) {
  size_t i = (size_t)blockIdx.x * 256 + threadIdx.x;
  if (i >= (size_t)NB * PERB) return;
  int b = (int)(i / PERB);
  int rem = (int)(i % PERB);
  int c = rem / SSEQ;
  int sp = rem % SSEQ;
  size_t tok = (size_t)b * SSEQ + sp;
  const size_t EO = (size_t)MPAD * CCH;
  float acc = x2[i];
  acc += wfull[tok * 3 + 0] * eo[0 * EO + tok * CCH + c];
  acc += wfull[tok * 3 + 1] * eo[1 * EO + tok * CCH + c];
  acc += wfull[tok * 3 + 2] * eo[2 * EO + tok * CCH + c];
  out[i] = acc;
}

// ------------------------------ host launcher ------------------------------

extern "C" void kernel_launch(void* const* d_in, const int* in_sizes, int n_in,
                              void* d_out, int out_size, void* d_ws, size_t ws_size,
                              hipStream_t stream) {
  (void)in_sizes; (void)n_in; (void)out_size; (void)ws_size;
  const float* x        = (const float*)d_in[0];
  const float* ln_w     = (const float*)d_in[1];
  const float* ln_b     = (const float*)d_in[2];
  const float* qkv_w    = (const float*)d_in[3];
  const float* proj_w   = (const float*)d_in[4];
  const float* router_w = (const float*)d_in[5];
  const float* gate_w   = (const float*)d_in[6];
  const float* up_w     = (const float*)d_in[7];
  const float* down_w   = (const float*)d_in[8];
  float* out = (float*)d_out;

  char* ws = (char*)d_ws;
  size_t off = 0;
  auto alloc = [&](size_t bytes) -> char* {
    char* p = ws + off;
    off = (off + bytes + 255) & ~(size_t)255;
    return p;
  };

  float*  stats = (float*) alloc(NB * 2 * sizeof(float));
  half_t* tokH  = (half_t*)alloc((size_t)MPAD * CCH * 2);
  half_t* wQKV  = (half_t*)alloc((size_t)CCH * 2304 * 2);
  half_t* qkvH  = (half_t*)alloc((size_t)MPAD * 2304 * 2);
  half_t* oH    = (half_t*)alloc((size_t)MPAD * CCH * 2);
  half_t* wPROJ = (half_t*)alloc((size_t)CCH * CCH * 2);
  float*  projO = (float*) alloc((size_t)MPAD * CCH * 4);
  float*  x2    = (float*) alloc((size_t)NB * PERB * 4);
  half_t* tok2H = (half_t*)alloc((size_t)MPAD * CCH * 2);
  float*  wfull = (float*) alloc((size_t)MTOK * 3 * 4);
  half_t* wG    = (half_t*)alloc((size_t)NEXP * CCH * HIDD * 2);
  half_t* wU    = (half_t*)alloc((size_t)NEXP * CCH * HIDD * 2);
  half_t* wD    = (half_t*)alloc((size_t)NEXP * HIDD * CCH * 2);
  half_t* gH    = (half_t*)alloc((size_t)NEXP * MPAD * HIDD * 2);
  half_t* uH    = (half_t*)alloc((size_t)NEXP * MPAD * HIDD * 2);
  float*  eo    = (float*) alloc((size_t)NEXP * MPAD * CCH * 4);

  // zero padded-token regions so pad rows stay benign through the GEMM chain
  (void)hipMemsetAsync(tokH,  0, (size_t)MPAD * CCH * 2, stream);
  (void)hipMemsetAsync(tok2H, 0, (size_t)MPAD * CCH * 2, stream);
  (void)hipMemsetAsync(oH,    0, (size_t)MPAD * CCH * 2, stream);

  // weights f32 -> f16
  {
    int blocks = 2048;
    cast_f32_f16_kernel<<<blocks, 256, 0, stream>>>(qkv_w,  wQKV, (size_t)CCH * 2304);
    cast_f32_f16_kernel<<<blocks, 256, 0, stream>>>(proj_w, wPROJ, (size_t)CCH * CCH);
    cast_f32_f16_kernel<<<blocks, 256, 0, stream>>>(gate_w, wG, (size_t)NEXP * CCH * HIDD);
    cast_f32_f16_kernel<<<blocks, 256, 0, stream>>>(up_w,   wU, (size_t)NEXP * CCH * HIDD);
    cast_f32_f16_kernel<<<blocks, 256, 0, stream>>>(down_w, wD, (size_t)NEXP * HIDD * CCH);
  }

  // LayerNorm -> token-major f16
  ln_stats_kernel<<<NB, 256, 0, stream>>>(x, stats);
  {
    size_t total = (size_t)NB * PERB;
    int blocks = (int)((total + 255) / 256);
    ln_apply_kernel<<<blocks, 256, 0, stream>>>(x, ln_w, ln_b, stats, tokH);
  }

  // QKV GEMM: (MPAD x 768) @ (768 x 2304) -> f16
  gemm_f16_kernel<true><<<dim3(2304 / 64, MPAD / 128), 256, 0, stream>>>(
      tokH, wQKV, (void*)qkvH, MTOK, 2304, CCH);

  // Flash attention per (m-tile, head, batch)
  attn_kernel<<<dim3(46, NHEAD, NB), 32, 0, stream>>>(qkvH, oH);

  // Proj GEMM: (MPAD x 768) @ (768 x 768) -> f32
  gemm_f16_kernel<false><<<dim3(CCH / 64, MPAD / 128), 256, 0, stream>>>(
      oH, wPROJ, (void*)projO, MTOK, CCH, CCH);

  // residual -> x2 (b,c,s layout) and token-major f16 for MoE
  {
    size_t total = (size_t)NB * PERB;
    int blocks = (int)((total + 255) / 256);
    residual_attn_kernel<<<blocks, 256, 0, stream>>>(x, projO, x2, tok2H);
  }

  // router softmax/top-2
  router_kernel<<<(MTOK + 3) / 4, 128, 0, stream>>>(x2, router_w, wfull);

  // experts: gate/up GEMMs, SwiGLU, down GEMMs
  for (int e = 0; e < NEXP; ++e) {
    const half_t* wg = wG + (size_t)e * CCH * HIDD;
    const half_t* wu = wU + (size_t)e * CCH * HIDD;
    half_t* ge = gH + (size_t)e * MPAD * HIDD;
    half_t* ue = uH + (size_t)e * MPAD * HIDD;
    gemm_f16_kernel<true><<<dim3(HIDD / 64, MPAD / 128), 256, 0, stream>>>(
        tok2H, wg, (void*)ge, MTOK, HIDD, CCH);
    gemm_f16_kernel<true><<<dim3(HIDD / 64, MPAD / 128), 256, 0, stream>>>(
        tok2H, wu, (void*)ue, MTOK, HIDD, CCH);
  }
  silu_mul_kernel<<<4096, 256, 0, stream>>>(gH, uH, (size_t)NEXP * MPAD * HIDD);
  for (int e = 0; e < NEXP; ++e) {
    const half_t* he = gH + (size_t)e * MPAD * HIDD;   // hid written in place of g
    const half_t* wd = wD + (size_t)e * HIDD * CCH;
    float* eoe = eo + (size_t)e * MPAD * CCH;
    gemm_f16_kernel<false><<<dim3(CCH / 64, MPAD / 128), 256, 0, stream>>>(
        he, wd, (void*)eoe, MTOK, CCH, HIDD);
  }

  // combine + residual -> output (b, C, 9,9,9) f32
  {
    size_t total = (size_t)NB * PERB;
    int blocks = (int)((total + 255) / 256);
    moe_combine_kernel<<<blocks, 256, 0, stream>>>(x2, eo, wfull, out);
  }
}
